// GAT_60971355734043
// MI455X (gfx1250) — matrix operand order repaired and checked by reading, not compiled, not run on hardware
//
#include <hip/hip_runtime.h>
#include <hip/hip_bf16.h>
#include <cstdint>

#define HC      128
#define NHEAD   4
#define CDIM    32
#define NSLOPE  0.2f
#define BROWS   64            // weight-tile columns staged in LDS per block
#define LDSPITCH 136          // 128 halves + 8-half (16B) TDM pad per row

typedef __attribute__((ext_vector_type(16))) _Float16 v16h;
typedef __attribute__((ext_vector_type(8)))  _Float16 v8h;
typedef __attribute__((ext_vector_type(8)))  float    v8f;
typedef __attribute__((ext_vector_type(4)))  unsigned int u32x4;
typedef __attribute__((ext_vector_type(8)))  int      i32x8;
typedef __attribute__((ext_vector_type(4)))  int      i32x4;

// ---------------- WMMA fragment loaders (per CDNA5 ISA 7.12.2 layouts) ----------------
// A 16-bit 16x32: lanes 0-15 hold row m, K {k0..k0+7} and {k0+16..k0+23};
//                 lanes 16-31 hold K {k0+8..k0+15} and {k0+24..k0+31}.
__device__ __forceinline__ v16h load_frag_a(const _Float16* __restrict__ row, int k0, int lanehi) {
    const _Float16* p = row + k0 + lanehi * 8;
    v8h lo = *(const v8h*)(p);
    v8h hi = *(const v8h*)(p + 16);
    v16h r;
#pragma unroll
    for (int i = 0; i < 8; ++i) { r[i] = lo[i]; r[i + 8] = hi[i]; }
    return r;
}

// B 16-bit 32x16 (from transposed weight Wt[n][k]): lanes 0-15 hold col n, K {k0..k0+7} and {k0+16..k0+23};
//                 lanes 16-31 hold K {k0+8..k0+15} and {k0+24..k0+31}.  Source here is LDS -> ds_load_b128.
__device__ __forceinline__ v16h load_frag_b(const _Float16* col, int k0, int lanehi) {
    const _Float16* p = col + k0 + lanehi * 8;
    v8h lo = *(const v8h*)(p);
    v8h hi = *(const v8h*)(p + 16);
    v16h r;
#pragma unroll
    for (int i = 0; i < 8; ++i) { r[i] = lo[i]; r[i + 8] = hi[i]; }
    return r;
}

// ---------------- GEMM: C[N x 128] = A16[N x 128] @ W (via Wt16), f32 accum ----------------
// block = 256 threads = 8 waves; wave -> 16 rows x 64 cols; grid = (ceil(N/128), 2)
// The 64x128 f16 weight tile is DMA'd into LDS by the Tensor Data Mover once per block.
__global__ void __launch_bounds__(256)
gat_gemm_f16(const _Float16* __restrict__ A, const _Float16* __restrict__ Bt,
             float* __restrict__ C, int nrows)
{
    __shared__ _Float16 Bs[BROWS * LDSPITCH];   // 17 KB, TDM-padded rows

    const int wave   = threadIdx.x >> 5;
    const int lane   = threadIdx.x & 31;
    const int lanehi = lane >> 4;
    const int lanelo = lane & 15;
    const int row0   = (blockIdx.x * 8 + wave) * 16;
    const int col0   = blockIdx.y * 64;

    // ---- TDM descriptor: 2-D tensor Bt[128][128] f16, tile = 64 rows x 128 elems ----
    if (wave == 0) {
        const unsigned lds = (unsigned)(size_t)(&Bs[0]);
        const unsigned long long ga =
            (unsigned long long)(size_t)(Bt + (size_t)col0 * HC);
        u32x4 g0;
        g0[0] = 1u;                                        // count=1 (valid), user mode
        g0[1] = lds;                                       // lds_addr
        g0[2] = (unsigned)(ga & 0xffffffffu);              // global_addr[31:0]
        g0[3] = (unsigned)((ga >> 32) & 0x01ffffffu)       // global_addr[56:32]
              | (2u << 30);                                // type=2 ("image")
        i32x8 g1;
        g1[0] = (int)((1u << 16)                           // data_size=1 -> 2 bytes
              |       (1u << 20)                           // pad_enable
              |       (5u << 22)                           // pad_interval: 64 DWORDs (256B row)
              |       (3u << 25));                         // pad_amount: 4 DWORDs (16B)
        g1[1] = (int)(128u << 16);                         // tensor_dim0 = 128 (bits 63:48)
        g1[2] = (int)(128u << 16);                         // tensor_dim1.lo = 128 (bits 95:80)
        g1[3] = (int)(128u << 16);                         // tile_dim0 = 128 (bits 127:112)
        g1[4] = 64;                                        // tile_dim1 = 64, tile_dim2 = 0
        g1[5] = 128;                                       // tensor_dim0_stride = 128 elems
        g1[6] = 0;
        g1[7] = 0;
        const i32x4 gz = {0, 0, 0, 0};
#if __clang_major__ >= 23
        const i32x8 gz8 = {0, 0, 0, 0, 0, 0, 0, 0};
        __builtin_amdgcn_tensor_load_to_lds(g0, g1, gz, gz, gz8, 0);
#else
        __builtin_amdgcn_tensor_load_to_lds(g0, g1, gz, gz, 0);
#endif
        __builtin_amdgcn_s_wait_tensorcnt((short)0);       // TDM done -> LDS valid
    }
    __syncthreads();

    int r = row0 + lanelo;
    if (r >= nrows) r = nrows - 1;          // clamp loads; EXEC stays all-ones for WMMA
    const _Float16* arow = A + (size_t)r * HC;

    v8f acc[4] = {};
#pragma unroll
    for (int k0 = 0; k0 < HC; k0 += 32) {
        v16h af = load_frag_a(arow, k0, lanehi);
#pragma unroll
        for (int t = 0; t < 4; ++t) {
            const _Float16* bcol = &Bs[(t * 16 + lanelo) * LDSPITCH];
            v16h bf = load_frag_b(bcol, k0, lanehi);
            acc[t] = __builtin_amdgcn_wmma_f32_16x16x32_f16(
                false, af, false, bf, (short)0, acc[t], false, false);
        }
    }

    const int mhi = lanehi * 8;             // C/D layout: VGPR i -> M = i (+8 for lanes 16-31)
#pragma unroll
    for (int t = 0; t < 4; ++t) {
        const int col = col0 + t * 16 + lanelo;
#pragma unroll
        for (int i = 0; i < 8; ++i) {
            const int row = row0 + mhi + i;
            if (row < nrows) C[(size_t)row * HC + col] = acc[t][i];
        }
    }
}

// ---------------- small helpers ----------------
__global__ void fill_f32(float* __restrict__ p, float v, size_t n) {
    size_t i = (size_t)blockIdx.x * blockDim.x + threadIdx.x;
    if (i < n) p[i] = v;
}

__global__ void cvt_f32_f16(const float* __restrict__ s, _Float16* __restrict__ d, size_t n) {
    size_t i = (size_t)blockIdx.x * blockDim.x + threadIdx.x;
    if (i < n) d[i] = (_Float16)s[i];
}

// Wt[n*128 + k] = (f16) W[k*128 + n]
__global__ void cvt_w_t_f16(const float* __restrict__ W, _Float16* __restrict__ Wt) {
    int i = blockIdx.x * blockDim.x + threadIdx.x;
    if (i >= HC * HC) return;
    int n = i >> 7, k = i & 127;
    Wt[i] = (_Float16)W[k * HC + n];
}

__device__ __forceinline__ void atomicMaxF32(float* addr, float val) {
    if (val >= 0.0f) atomicMax((int*)addr, __float_as_int(val));
    else             atomicMin((unsigned int*)addr, __float_as_uint(val));
}

// ---------------- edge phase ----------------
// wave per edge: e[eid][h] = sum_c lrelu(xl[s][h,c]+xr[d][h,c]) * att[h,c]; atomic segment max
__global__ void __launch_bounds__(256)
edge_score(const float* __restrict__ xl, const float* __restrict__ xr,
           const long long* __restrict__ srcA, const long long* __restrict__ dstA,
           int E_, int Nn, const float* __restrict__ att,
           float* __restrict__ ebuf, float* __restrict__ mbuf)
{
    const int eid  = blockIdx.x * (blockDim.x >> 5) + (threadIdx.x >> 5);
    const int lane = threadIdx.x & 31;
    const int Etot = E_ + Nn;
    if (eid >= Etot) return;
    int s, d;
    if (eid < E_) { s = (int)srcA[eid]; d = (int)dstA[eid]; }
    else          { s = d = eid - E_; }

    const float* pl = xl + (size_t)s * HC;
    const float* pr = xr + (size_t)d * HC;
    float v[NHEAD];
#pragma unroll
    for (int h = 0; h < NHEAD; ++h) {
        const int i = h * CDIM + lane;
        float z  = pl[i] + pr[i];
        float lr = z > 0.0f ? z : NSLOPE * z;
        v[h] = lr * att[i];
    }
#pragma unroll
    for (int off = 16; off >= 1; off >>= 1)
#pragma unroll
        for (int h = 0; h < NHEAD; ++h) v[h] += __shfl_xor(v[h], off, 32);

    if (lane == 0) {
#pragma unroll
        for (int h = 0; h < NHEAD; ++h) {
            ebuf[(size_t)eid * NHEAD + h] = v[h];
            atomicMaxF32(&mbuf[(size_t)d * NHEAD + h], v[h]);
        }
    }
}

// thread per (edge, head): a = exp(e - m[dst]); denom[dst] += a
__global__ void edge_exp(float* __restrict__ ebuf, const float* __restrict__ mbuf,
                         float* __restrict__ dbuf, const long long* __restrict__ dstA,
                         int E_, int Nn)
{
    const long long tot = (long long)(E_ + Nn) * NHEAD;
    const long long t   = (long long)blockIdx.x * blockDim.x + threadIdx.x;
    if (t >= tot) return;
    const int eid = (int)(t >> 2), h = (int)(t & 3);
    const int d   = (eid < E_) ? (int)dstA[eid] : eid - E_;
    const float a = __expf(ebuf[t] - mbuf[(size_t)d * NHEAD + h]);
    ebuf[t] = a;
    atomicAdd(&dbuf[(size_t)d * NHEAD + h], a);
}

// wave per edge: acc[dst] += (a/denom[dst]) * xl[src]   (L2-resident f32 atomics)
__global__ void __launch_bounds__(256)
edge_aggr(const float* __restrict__ ebuf, const float* __restrict__ dbuf,
          const float* __restrict__ xl,
          const long long* __restrict__ srcA, const long long* __restrict__ dstA,
          int E_, int Nn, float* __restrict__ acc)
{
    const int eid  = blockIdx.x * (blockDim.x >> 5) + (threadIdx.x >> 5);
    const int lane = threadIdx.x & 31;
    const int Etot = E_ + Nn;
    if (eid >= Etot) return;
    int s, d;
    if (eid < E_) { s = (int)srcA[eid]; d = (int)dstA[eid]; }
    else          { s = d = eid - E_; }

    const float* pl = xl + (size_t)s * HC;
    float*       pa = acc + (size_t)d * HC;
#pragma unroll
    for (int h = 0; h < NHEAD; ++h) {
        const float alpha = ebuf[(size_t)eid * NHEAD + h] / dbuf[(size_t)d * NHEAD + h];
        const int i = h * CDIM + lane;
        atomicAdd(&pa[i], alpha * pl[i]);
    }
}

__global__ void finalize_relu(const float* __restrict__ acc, const float* __restrict__ bias,
                              float* __restrict__ hout, size_t n)
{
    size_t i = (size_t)blockIdx.x * blockDim.x + threadIdx.x;
    if (i >= n) return;
    float v = acc[i] + bias[i & (HC - 1)];
    hout[i] = v > 0.0f ? v : 0.0f;
}

// wave per node: out[n] = h[n] . Wf + bf
__global__ void __launch_bounds__(256)
fc_final(const float* __restrict__ h, const float* __restrict__ Wf,
         const float* __restrict__ bf, float* __restrict__ out, int Nn)
{
    const int node = blockIdx.x * (blockDim.x >> 5) + (threadIdx.x >> 5);
    const int lane = threadIdx.x & 31;
    if (node >= Nn) return;
    const float* row = h + (size_t)node * HC;
    float s = 0.0f;
#pragma unroll
    for (int j = 0; j < 4; ++j) { const int i = j * 32 + lane; s += row[i] * Wf[i]; }
#pragma unroll
    for (int off = 16; off >= 1; off >>= 1) s += __shfl_xor(s, off, 32);
    if (lane == 0) out[node] = s + bf[0];
}

// ---------------- host orchestration ----------------
extern "C" void kernel_launch(void* const* d_in, const int* in_sizes, int n_in,
                              void* d_out, int out_size, void* d_ws, size_t ws_size,
                              hipStream_t stream)
{
    (void)n_in; (void)out_size; (void)ws_size;
    const float*     x  = (const float*)d_in[0];
    const long long* ei = (const long long*)d_in[1];
    const int N    = in_sizes[0] / HC;
    const int E    = in_sizes[1] / 2;
    const int Etot = E + N;
    const long long* srcA = ei;
    const long long* dstA = ei + E;

    char* w = (char*)d_ws;
    auto carve = [&](size_t bytes) -> void* {
        void* p = (void*)w; w += (bytes + 255) & ~(size_t)255; return p;
    };
    const size_t nElem = (size_t)N * HC;
    float*    xl   = (float*)carve(nElem * sizeof(float));
    float*    xr   = (float*)carve(nElem * sizeof(float));
    float*    acc  = (float*)carve(nElem * sizeof(float));
    float*    hA   = (float*)carve(nElem * sizeof(float));
    float*    hB   = (float*)carve(nElem * sizeof(float));
    float*    ebuf = (float*)carve((size_t)Etot * NHEAD * sizeof(float));
    float*    mbuf = (float*)carve((size_t)N * NHEAD * sizeof(float));
    float*    dbuf = (float*)carve((size_t)N * NHEAD * sizeof(float));
    _Float16* a16  = (_Float16*)carve(nElem * sizeof(_Float16));
    _Float16* wl16 = (_Float16*)carve((size_t)HC * HC * sizeof(_Float16));
    _Float16* wr16 = (_Float16*)carve((size_t)HC * HC * sizeof(_Float16));

    auto layer = [&](const float* hin, const float* Wl, const float* Wr,
                     const float* att, const float* bias, float* hout) {
        cvt_f32_f16<<<(int)((nElem + 255) / 256), 256, 0, stream>>>(hin, a16, nElem);
        cvt_w_t_f16<<<(HC * HC + 255) / 256, 256, 0, stream>>>(Wl, wl16);
        cvt_w_t_f16<<<(HC * HC + 255) / 256, 256, 0, stream>>>(Wr, wr16);

        dim3 g((N + 127) / 128, 2);
        gat_gemm_f16<<<g, 256, 0, stream>>>(a16, wl16, xl, N);
        gat_gemm_f16<<<g, 256, 0, stream>>>(a16, wr16, xr, N);

        const size_t nh = (size_t)N * NHEAD;
        fill_f32<<<(int)((nh + 255) / 256), 256, 0, stream>>>(mbuf, -__builtin_inff(), nh);
        fill_f32<<<(int)((nh + 255) / 256), 256, 0, stream>>>(dbuf, 0.0f, nh);
        fill_f32<<<(int)((nElem + 255) / 256), 256, 0, stream>>>(acc, 0.0f, nElem);

        edge_score<<<(Etot + 7) / 8, 256, 0, stream>>>(xl, xr, srcA, dstA, E, N, att, ebuf, mbuf);
        const long long nt = (long long)Etot * NHEAD;
        edge_exp<<<(int)((nt + 255) / 256), 256, 0, stream>>>(ebuf, mbuf, dbuf, dstA, E, N);
        edge_aggr<<<(Etot + 7) / 8, 256, 0, stream>>>(ebuf, dbuf, xl, srcA, dstA, E, N, acc);

        finalize_relu<<<(int)((nElem + 255) / 256), 256, 0, stream>>>(acc, bias, hout, nElem);
    };

    layer(x,  (const float*)d_in[2],  (const float*)d_in[3],  (const float*)d_in[4],  (const float*)d_in[5],  hA);
    layer(hA, (const float*)d_in[6],  (const float*)d_in[7],  (const float*)d_in[8],  (const float*)d_in[9],  hB);
    layer(hB, (const float*)d_in[10], (const float*)d_in[11], (const float*)d_in[12], (const float*)d_in[13], hA);

    fc_final<<<(N + 7) / 8, 256, 0, stream>>>(hA, (const float*)d_in[14],
                                              (const float*)d_in[15], (float*)d_out, N);
}
